// SimplexAttention_88691074662704
// MI455X (gfx1250) — matrix operand order repaired and checked by reading, not compiled, not run on hardware
//
#include <hip/hip_runtime.h>
#include <hip/hip_bf16.h>

// ---------------- problem constants ----------------
#define NB   8
#define NS   2048
#define DD   256

typedef __attribute__((ext_vector_type(16))) __bf16       v16bf;
typedef __attribute__((ext_vector_type(8)))  float        v8f;
typedef __attribute__((ext_vector_type(4)))  unsigned int u32x4;

union BfOp { v16bf v; u32x4 u[2]; };

__device__ __forceinline__ __bf16 f2bf(float f) {
  return __builtin_bit_cast(__bf16, __float2bfloat16(f));
}

__device__ __forceinline__ v8f wmma_bf16(v16bf a, v16bf b, v8f c) {
  // D = A(16x32 bf16) * B(32x16 bf16) + C(16x16 f32)
  return __builtin_amdgcn_wmma_f32_16x16x32_bf16(
      /*neg_a=*/false, a, /*neg_b=*/false, b,
      /*c_mod=*/(short)0, c, /*reuse_a=*/false, /*reuse_b=*/false);
}

// ---------------------------------------------------------------------------
// Causal conv (K=3, replicate-left pad):  y[b,n,o] = bias[o] +
//   sum_t sum_i x[b, max(n-2+t,0), i] * w[o,i,t]
// One block = 16 output rows, 256 threads = one output channel each.
// Emits bf16 output + fp32 squared row norms.
// ---------------------------------------------------------------------------
__global__ __launch_bounds__(256) void conv_kernel(
    const float* __restrict__ x, const float* __restrict__ w,
    const float* __restrict__ bias, __bf16* __restrict__ ybf,
    float* __restrict__ ynorm)
{
  __shared__ float xs[18][DD];     // rows n0-2 .. n0+15
  __shared__ float part[16][8];

  const int tid = threadIdx.x;
  const int b   = blockIdx.x / (NS / 16);
  const int n0  = (blockIdx.x % (NS / 16)) * 16;

  #pragma unroll
  for (int rr = 0; rr < 18; ++rr) {
    int row = n0 - 2 + rr;
    row = row < 0 ? 0 : row;
    xs[rr][tid] = x[((size_t)b * NS + row) * DD + tid];
  }
  __syncthreads();

  float acc[16];
  const float bo = bias[tid];
  #pragma unroll
  for (int n = 0; n < 16; ++n) acc[n] = bo;

  const float* wp = w + (size_t)tid * DD * 3;   // w[o, :, :]
  for (int i = 0; i < DD; ++i) {
    const float w0 = wp[i * 3 + 0];
    const float w1 = wp[i * 3 + 1];
    const float w2 = wp[i * 3 + 2];
    #pragma unroll
    for (int n = 0; n < 16; ++n)
      acc[n] = fmaf(xs[n][i], w0, fmaf(xs[n + 1][i], w1, fmaf(xs[n + 2][i], w2, acc[n])));
  }

  const int lane = tid & 31, wv = tid >> 5;
  #pragma unroll
  for (int n = 0; n < 16; ++n) {
    ybf[((size_t)b * NS + n0 + n) * DD + tid] = f2bf(acc[n]);
    float v = acc[n] * acc[n];
    #pragma unroll
    for (int m = 1; m <= 16; m <<= 1) v += __shfl_xor(v, m, 32);
    if (lane == 0) part[n][wv] = v;
  }
  __syncthreads();
  if (tid < 16) {
    float s = 0.f;
    #pragma unroll
    for (int q = 0; q < 8; ++q) s += part[tid][q];
    ynorm[(size_t)b * NS + n0 + tid] = s;
  }
}

// ---------------------------------------------------------------------------
// vt[b][d][n] = (n > 0) ? vals[b][n-1][d] : 0   (bf16, transposed layout)
// 32x32 tile transpose through LDS.
// ---------------------------------------------------------------------------
__global__ __launch_bounds__(256) void shiftT_kernel(
    const float* __restrict__ vals, __bf16* __restrict__ vt)
{
  __shared__ __bf16 ls[32][33];
  const int bid = blockIdx.x;
  const int dt = bid % (DD / 32);
  const int nt = (bid / (DD / 32)) % (NS / 32);
  const int b  = bid / ((DD / 32) * (NS / 32));
  const int n0 = nt * 32, d0 = dt * 32;
  const int c = threadIdx.x & 31;
  const int r = threadIdx.x >> 5;

  #pragma unroll
  for (int rr = r; rr < 32; rr += 8) {
    const int n = n0 + rr;
    const float v = (n > 0) ? vals[((size_t)b * NS + n - 1) * DD + d0 + c] : 0.f;
    ls[rr][c] = f2bf(v);
  }
  __syncthreads();
  #pragma unroll
  for (int rr = r; rr < 32; rr += 8)
    vt[((size_t)b * DD + d0 + rr) * NS + n0 + c] = ls[c][rr];
}

// ---------------------------------------------------------------------------
// Flash attention with Euclidean-distance scores.
//   score(i,j) = -sqrt(max(q2[i] + k2[j] - 2 q.k, 0)), causal j<=i, softmax,
//   O = P @ V_shift.
// Block: 128 threads = 4 waves; wave owns 16 query rows. j tiles of 32.
// ---------------------------------------------------------------------------
__global__ __launch_bounds__(128) void attn_kernel(
    const __bf16* __restrict__ qc, const __bf16* __restrict__ kc,
    const __bf16* __restrict__ vt, const float* __restrict__ q2,
    const float* __restrict__ k2, float* __restrict__ out)
{
  __shared__ __align__(16) __bf16 ksh[32 * DD];      // K tile, row-major [32][256]
  __shared__ __align__(16) __bf16 vsh[DD * 32];      // V^T tile [256][32]
  __shared__ __align__(16) __bf16 psh[4 * 16 * 32];  // per-wave P scratch

  const int tid  = threadIdx.x;
  const int wave = tid >> 5;
  const int lane = tid & 31;
  const int nlo  = lane & 15;
  const int kh   = lane >> 4;

  const int b  = blockIdx.x / (NS / 64);
  const int rb = blockIdx.x % (NS / 64);
  const int i0 = rb * 64 + wave * 16;     // wave's first query row
  const int my_last = i0 + 15;

  // ---- preload Q as WMMA A-operand (8 chunks of K=32) ----
  const __bf16* qrow = qc + ((size_t)b * NS + (i0 + nlo)) * DD;
  v16bf qa[8];
  #pragma unroll
  for (int c = 0; c < 8; ++c) {
    BfOp t;
    t.u[0] = *(const u32x4*)(qrow + 32 * c + 8 * kh);
    t.u[1] = *(const u32x4*)(qrow + 32 * c + 16 + 8 * kh);
    qa[c] = t.v;
  }

  // per-lane row state: rows i0 + 8*kh + r, replicated across 16-lane half
  float m_i[8], l_i[8], q2r[8];
  #pragma unroll
  for (int r = 0; r < 8; ++r) {
    q2r[r] = q2[(size_t)b * NS + i0 + 8 * kh + r];
    m_i[r] = -3.0e38f;
    l_i[r] = 0.f;
  }

  const v8f vzero = {0.f, 0.f, 0.f, 0.f, 0.f, 0.f, 0.f, 0.f};
  v8f o_acc[16];
  #pragma unroll
  for (int t = 0; t < 16; ++t) o_acc[t] = vzero;

  const int ntiles = (rb * 64 + 63) / 32 + 1;   // cover causal range of block

  for (int jt = 0; jt < ntiles; ++jt) {
    const int j0 = jt * 32;
    __syncthreads();  // previous-iteration LDS reads done

    // ---- cooperative tile loads: K tile is contiguous 16 KB ----
    {
      const u32x4* gk = (const u32x4*)(kc + ((size_t)b * NS + j0) * DD);
      u32x4* sk = (u32x4*)ksh;
      #pragma unroll
      for (int p = 0; p < 8; ++p) sk[tid + 128 * p] = gk[tid + 128 * p];
      u32x4* sv = (u32x4*)vsh;
      #pragma unroll
      for (int p = 0; p < 8; ++p) {
        const int q = tid + 128 * p;
        const int d = q >> 2, jl = (q & 3) * 8;
        sv[q] = *(const u32x4*)(vt + ((size_t)b * DD + d) * NS + j0 + jl);
      }
    }
    __syncthreads();

    if (jt + 1 < ntiles)   // pull next K tile toward the caches
      __builtin_prefetch(kc + ((size_t)b * NS + j0 + 32) * DD + (size_t)tid * 64, 0, 1);

    if (j0 > my_last) continue;   // wave past its causal range (barriers balanced)

    // ---- S = Q K^T over D=256 (two 16-col halves of the 32-wide j tile) ----
    v8f s0 = vzero, s1 = vzero;
    #pragma unroll
    for (int c = 0; c < 8; ++c) {
      BfOp b0, b1;
      const __bf16* kr0 = ksh + (size_t)nlo * DD + 32 * c + 8 * kh;
      const __bf16* kr1 = ksh + (size_t)(16 + nlo) * DD + 32 * c + 8 * kh;
      b0.u[0] = *(const u32x4*)(kr0); b0.u[1] = *(const u32x4*)(kr0 + 16);
      b1.u[0] = *(const u32x4*)(kr1); b1.u[1] = *(const u32x4*)(kr1 + 16);
      s0 = wmma_bf16(qa[c], b0.v, s0);
      s1 = wmma_bf16(qa[c], b1.v, s1);
    }

    // ---- distances + causal mask + online softmax ----
    const float k2a = k2[(size_t)b * NS + j0 + nlo];
    const float k2b = k2[(size_t)b * NS + j0 + 16 + nlo];
    float p0[8], p1[8], tm[8];
    #pragma unroll
    for (int r = 0; r < 8; ++r) {
      const int i  = i0 + 8 * kh + r;
      const int ja = j0 + nlo, jb = j0 + 16 + nlo;
      float d2a = fmaxf(q2r[r] + k2a - 2.0f * s0[r], 0.f);
      float d2b = fmaxf(q2r[r] + k2b - 2.0f * s1[r], 0.f);
      const float sa = (ja <= i) ? -__builtin_sqrtf(d2a) : -3.0e38f;
      const float sb = (jb <= i) ? -__builtin_sqrtf(d2b) : -3.0e38f;
      p0[r] = sa; p1[r] = sb;
      tm[r] = fmaxf(sa, sb);
    }
    #pragma unroll
    for (int r = 0; r < 8; ++r) {
      #pragma unroll
      for (int m = 1; m <= 8; m <<= 1)          // stays within 16-lane half
        tm[r] = fmaxf(tm[r], __shfl_xor(tm[r], m, 32));
    }
    float scale[8];
    #pragma unroll
    for (int r = 0; r < 8; ++r) {
      const float mn = fmaxf(m_i[r], tm[r]);
      scale[r] = __expf(m_i[r] - mn);
      m_i[r]   = mn;
      const float pa = __expf(p0[r] - mn);
      const float pb = __expf(p1[r] - mn);
      p0[r] = pa; p1[r] = pb;
      float rs = pa + pb;
      #pragma unroll
      for (int m = 1; m <= 8; m <<= 1) rs += __shfl_xor(rs, m, 32);
      l_i[r] = l_i[r] * scale[r] + rs;
    }
    #pragma unroll
    for (int t = 0; t < 16; ++t)
      #pragma unroll
      for (int r = 0; r < 8; ++r) o_acc[t][r] *= scale[r];

    // ---- C-layout P -> A-layout via per-wave LDS (in-order within wave) ----
    __bf16* pw = psh + wave * (16 * 32);
    #pragma unroll
    for (int r = 0; r < 8; ++r) {
      pw[(8 * kh + r) * 32 + nlo]      = f2bf(p0[r]);
      pw[(8 * kh + r) * 32 + 16 + nlo] = f2bf(p1[r]);
    }
    BfOp pt;
    pt.u[0] = *(const u32x4*)(pw + (size_t)nlo * 32 + 8 * kh);
    pt.u[1] = *(const u32x4*)(pw + (size_t)nlo * 32 + 16 + 8 * kh);
    const v16bf pmat = pt.v;

    // ---- O += P @ V  (16 output d-chunks of 16) ----
    #pragma unroll
    for (int t = 0; t < 16; ++t) {
      BfOp bv;
      const __bf16* vr = vsh + (size_t)(16 * t + nlo) * 32 + 8 * kh;
      bv.u[0] = *(const u32x4*)(vr);
      bv.u[1] = *(const u32x4*)(vr + 16);
      o_acc[t] = wmma_bf16(pmat, bv.v, o_acc[t]);
    }
  }

  // ---- normalize & store fp32 ----
  float inv[8];
  #pragma unroll
  for (int r = 0; r < 8; ++r) inv[r] = 1.0f / l_i[r];
  #pragma unroll
  for (int t = 0; t < 16; ++t)
    #pragma unroll
    for (int r = 0; r < 8; ++r)
      out[((size_t)b * NS + i0 + 8 * kh + r) * DD + 16 * t + nlo] = o_acc[t][r] * inv[r];
}

// ---------------------------------------------------------------------------
extern "C" void kernel_launch(void* const* d_in, const int* in_sizes, int n_in,
                              void* d_out, int out_size, void* d_ws, size_t ws_size,
                              hipStream_t stream) {
  (void)in_sizes; (void)n_in; (void)out_size; (void)ws_size;
  const float* query  = (const float*)d_in[0];
  const float* keys   = (const float*)d_in[1];
  const float* vals   = (const float*)d_in[2];
  // d_in[3] = mask (tril) — causality computed from indices instead
  const float* conv_w = (const float*)d_in[4];
  const float* conv_b = (const float*)d_in[5];
  float* out = (float*)d_out;

  const size_t nElem = (size_t)NB * NS * DD;
  char* ws = (char*)d_ws;
  __bf16* qc = (__bf16*)ws;  ws += nElem * 2;
  __bf16* kc = (__bf16*)ws;  ws += nElem * 2;
  __bf16* vt = (__bf16*)ws;  ws += nElem * 2;
  float*  q2 = (float*)ws;   ws += (size_t)NB * NS * 4;
  float*  k2 = (float*)ws;

  conv_kernel<<<NB * (NS / 16), 256, 0, stream>>>(query, conv_w, conv_b, qc, q2);
  conv_kernel<<<NB * (NS / 16), 256, 0, stream>>>(keys,  conv_w, conv_b, kc, k2);
  shiftT_kernel<<<NB * (NS / 32) * (DD / 32), 256, 0, stream>>>(vals, vt);
  attn_kernel<<<NB * (NS / 64), 128, 0, stream>>>(qc, kc, vt, q2, k2, out);
}
